// MyLSTM_38405597561655
// MI455X (gfx1250) — matrix operand order repaired and checked by reading, not compiled, not run on hardware
//
#include <hip/hip_runtime.h>
#include <hip/hip_bf16.h>

// ---------------------------------------------------------------------------
// LSTM on gfx1250 (MI455X): bf16 WMMA GEMMs (v_wmma_f32_16x16x32_bf16),
// f32 accumulate, software-pipelined fragment loads (register double-buffer),
// non-temporal stores for the write-once streams (pregates P, outs).
// ---------------------------------------------------------------------------

typedef __bf16 bf16_t;
typedef __attribute__((ext_vector_type(16))) __bf16 v16bf;
typedef __attribute__((ext_vector_type(8)))  float  v8f;
typedef __attribute__((ext_vector_type(4)))  unsigned int u32x4;  // native vec

#define LSTM_B 128
#define LSTM_T 512
#define LSTM_D 768
#define LSTM_H 768
#define NTOT   (4 * LSTM_H)          // 3072 concatenated gate columns
#define MTOT   (LSTM_T * LSTM_B)     // 65536 rows of the input projection

union Frag {                          // 16 bf16 = one WMMA A or B operand
    v16bf v;
    uint4 u[2];
};

union Pack8 {                         // 8 bf16 = 16 bytes
    bf16_t h[8];
    u32x4  u;
};

__device__ __forceinline__ float sigm(float x) {
    return 1.0f / (1.0f + __expf(-x));
}

__device__ __forceinline__ void loadA(Frag& f, const bf16_t* aptr, int k0) {
    f.u[0] = *(const uint4*)(aptr + k0);        // K = k0+abase    .. +7
    f.u[1] = *(const uint4*)(aptr + k0 + 16);   // K = k0+abase+16 .. +23
}

__device__ __forceinline__ void loadB(Frag& f, const bf16_t* brow, int k0) {
    f.u[0] = *(const uint4*)(brow + k0);        // K = k0+kb    .. +7 (transposed W)
    f.u[1] = *(const uint4*)(brow + k0 + 16);   // K = k0+kb+16 .. +23
}

// ---------------------------------------------------------------------------
// Pack x [B,T,D] f32  ->  Xb [T*B, D] bf16   (row m = t*B + b)
// ---------------------------------------------------------------------------
__global__ __launch_bounds__(256) void pack_x_kernel(
    const float* __restrict__ x, bf16_t* __restrict__ Xb)
{
    size_t tid = (size_t)blockIdx.x * blockDim.x + threadIdx.x;
    const size_t nthr = (size_t)MTOT * (LSTM_D / 4);
    if (tid >= nthr) return;
    const int m  = (int)(tid / (LSTM_D / 4));
    const int dq = (int)(tid % (LSTM_D / 4)) * 4;
    const int t  = m >> 7;            // / B
    const int b  = m & 127;           // % B
    const float4 src = *(const float4*)(x + ((size_t)b * LSTM_T + t) * LSTM_D + dq);
    Pack8 p;  // use first 4
    p.h[0] = (bf16_t)src.x; p.h[1] = (bf16_t)src.y;
    p.h[2] = (bf16_t)src.z; p.h[3] = (bf16_t)src.w;
    *(uint2*)(Xb + (size_t)m * LSTM_D + dq) = *(const uint2*)&p.u;
}

// ---------------------------------------------------------------------------
// Pack 4 gate weight matrices [D,H] f32 (row-major [k][n]) into transposed
// bf16 Wp [4*H][D]:  Wp[(g*H + n)*D + k] = W_g[k*H + n]
// ---------------------------------------------------------------------------
__global__ __launch_bounds__(256) void pack_w_kernel(
    const float* __restrict__ W0, const float* __restrict__ W1,
    const float* __restrict__ W2, const float* __restrict__ W3,
    bf16_t* __restrict__ Wp, int K)
{
    size_t tid = (size_t)blockIdx.x * blockDim.x + threadIdx.x;
    const size_t total = (size_t)4 * LSTM_H * K;
    if (tid >= total) return;
    const int g = (int)(tid / ((size_t)LSTM_H * K));
    const size_t q = tid % ((size_t)LSTM_H * K);
    const int n = (int)(q / K);
    const int k = (int)(q % K);
    const float* W = (g == 0) ? W0 : (g == 1) ? W1 : (g == 2) ? W2 : W3;
    Wp[tid] = (bf16_t)W[(size_t)k * LSTM_H + n];
}

// ---------------------------------------------------------------------------
// Pack 4 bias vectors [H] into one f32 [4*H]
// ---------------------------------------------------------------------------
__global__ __launch_bounds__(256) void pack_b_kernel(
    const float* __restrict__ b0, const float* __restrict__ b1,
    const float* __restrict__ b2, const float* __restrict__ b3,
    float* __restrict__ bp)
{
    int tid = blockIdx.x * blockDim.x + threadIdx.x;
    if (tid >= NTOT) return;
    const int g = tid / LSTM_H;
    const int n = tid % LSTM_H;
    const float* b = (g == 0) ? b0 : (g == 1) ? b1 : (g == 2) ? b2 : b3;
    bp[tid] = b[n];
}

// ---------------------------------------------------------------------------
// Zero initial state: h0 (bf16) and c (f32), both B*H elements
// ---------------------------------------------------------------------------
__global__ __launch_bounds__(256) void init_state_kernel(
    bf16_t* __restrict__ h0, float* __restrict__ c)
{
    int tid = blockIdx.x * blockDim.x + threadIdx.x;
    if (tid >= LSTM_B * LSTM_H) return;
    h0[tid] = (bf16_t)0.0f;
    c[tid]  = 0.0f;
}

// ---------------------------------------------------------------------------
// Phase 1: pregates P = Xb @ Wxp + bx, stored bf16 transposed:
//   P[((t*4 + g)*H + n)*B + b]
// Each wave computes a 16(M) x 64(N) strip: 4 WMMA tiles sharing the A frag.
// Software pipeline: loads for k-step k+1 issued before WMMAs of k-step k.
// ---------------------------------------------------------------------------
__global__ __launch_bounds__(256) void xproj_kernel(
    const bf16_t* __restrict__ Xb,    // [MTOT, D]
    const bf16_t* __restrict__ Wxp,   // [NTOT, D] (transposed)
    const float*  __restrict__ bxp,   // [NTOT]
    bf16_t* __restrict__ P)
{
    const int lane = threadIdx.x & 31;
    const int wid  = (int)((blockIdx.x * blockDim.x + threadIdx.x) >> 5);
    const int nStrips = NTOT / 64;                 // 48
    const int strip = wid % nStrips;
    const int tileM = wid / nStrips;
    if (tileM >= MTOT / 16) return;

    const int mbase  = tileM << 4;
    const int nstrip = strip << 6;
    const int l15  = lane & 15;
    const int hi16 = lane >> 4;                    // 0 | 1
    const int abase = hi16 << 3;                   // A: K offset 0 | 8
    const int kb    = hi16 << 3;                   // B: K offset 0 | 8

    const bf16_t* aptr = Xb + (size_t)(mbase + l15) * LSTM_D + abase;
    const bf16_t* bptr[4];
#pragma unroll
    for (int i = 0; i < 4; ++i)
        bptr[i] = Wxp + (size_t)(nstrip + (i << 4) + l15) * LSTM_D + kb;

    v8f acc[4];
#pragma unroll
    for (int i = 0; i < 4; ++i) acc[i] = 0.0f;

    // Pipelined main loop: two fragment buffers, 64 K per iteration.
    Frag a0, a1, b0[4], b1[4];
    loadA(a0, aptr, 0);
#pragma unroll
    for (int i = 0; i < 4; ++i) loadB(b0[i], bptr[i], 0);

#pragma unroll 3
    for (int k0 = 0; k0 < LSTM_D; k0 += 64) {
        // stage 1: prefetch odd k-step, then consume even k-step
        loadA(a1, aptr, k0 + 32);
#pragma unroll
        for (int i = 0; i < 4; ++i) loadB(b1[i], bptr[i], k0 + 32);
#pragma unroll
        for (int i = 0; i < 4; ++i)
            acc[i] = __builtin_amdgcn_wmma_f32_16x16x32_bf16(
                false, a0.v, false, b0[i].v, (short)0, acc[i], false, false);
        // stage 2: prefetch next even k-step, then consume odd k-step
        if (k0 + 64 < LSTM_D) {
            loadA(a0, aptr, k0 + 64);
#pragma unroll
            for (int i = 0; i < 4; ++i) loadB(b0[i], bptr[i], k0 + 64);
        }
#pragma unroll
        for (int i = 0; i < 4; ++i)
            acc[i] = __builtin_amdgcn_wmma_f32_16x16x32_bf16(
                false, a1.v, false, b1[i].v, (short)0, acc[i], false, false);
    }

    // Store: lane owns 8 consecutive b values (rows) at a fixed column n.
    const int t  = mbase >> 7;                      // / B
    const int b0r = (mbase & 127) + (hi16 << 3);
#pragma unroll
    for (int i = 0; i < 4; ++i) {
        const int ntot = nstrip + (i << 4) + l15;
        const int g = ntot / LSTM_H;
        const int n = ntot - g * LSTM_H;
        const float bias = bxp[ntot];
        Pack8 p;
#pragma unroll
        for (int r = 0; r < 8; ++r) p.h[r] = (bf16_t)(acc[i][r] + bias);
        const size_t idx = ((size_t)((t * 4 + g) * LSTM_H + n)) * LSTM_B + b0r;
        __builtin_nontemporal_store(p.u, (u32x4*)(P + idx));   // write-once stream
    }
}

// ---------------------------------------------------------------------------
// Phase 2: one timestep.  Gates = sigma/tanh(P_t + h_prev @ Whp + bh).
// Each wave computes one 16(b) x 16(n) tile for ALL FOUR gates (4 WMMAs per
// k-step sharing one A fragment), then does the LSTM cell update in regs.
// Same register double-buffer pipeline as phase 1.
// ---------------------------------------------------------------------------
__global__ __launch_bounds__(256) void lstm_step_kernel(
    const bf16_t* __restrict__ hprev,  // [B, H] bf16
    const bf16_t* __restrict__ Whp,    // [NTOT, H] (transposed)
    const float*  __restrict__ bhp,    // [NTOT]
    const bf16_t* __restrict__ P,      // [T][4][H][B]
    float*  __restrict__ c_ws,         // [H][B]
    bf16_t* __restrict__ hnext,        // [B, H] bf16
    float*  __restrict__ out,          // full d_out
    int t)
{
    const int lane = threadIdx.x & 31;
    const int wid  = (int)((blockIdx.x * blockDim.x + threadIdx.x) >> 5);
    const int tileM = wid & 7;                      // 8 tiles of 16 rows (b)
    const int tileN = wid >> 3;                     // 48 tiles of 16 cols (n)
    if (tileN >= LSTM_H / 16) return;

    const int l15  = lane & 15;
    const int hi16 = lane >> 4;
    const int abase = hi16 << 3;
    const int kb    = hi16 << 3;
    const int btile = tileM << 4;
    const int n  = (tileN << 4) + l15;

    const bf16_t* aptr = hprev + (size_t)(btile + l15) * LSTM_H + abase;
    const bf16_t* bptr[4];
#pragma unroll
    for (int g = 0; g < 4; ++g)
        bptr[g] = Whp + (size_t)(g * LSTM_H + n) * LSTM_H + kb;

    v8f acc[4];
#pragma unroll
    for (int g = 0; g < 4; ++g) acc[g] = 0.0f;

    Frag a0, a1, b0[4], b1[4];
    loadA(a0, aptr, 0);
#pragma unroll
    for (int g = 0; g < 4; ++g) loadB(b0[g], bptr[g], 0);

#pragma unroll 3
    for (int k0 = 0; k0 < LSTM_H; k0 += 64) {
        loadA(a1, aptr, k0 + 32);
#pragma unroll
        for (int g = 0; g < 4; ++g) loadB(b1[g], bptr[g], k0 + 32);
#pragma unroll
        for (int g = 0; g < 4; ++g)
            acc[g] = __builtin_amdgcn_wmma_f32_16x16x32_bf16(
                false, a0.v, false, b0[g].v, (short)0, acc[g], false, false);
        if (k0 + 64 < LSTM_H) {
            loadA(a0, aptr, k0 + 64);
#pragma unroll
            for (int g = 0; g < 4; ++g) loadB(b0[g], bptr[g], k0 + 64);
        }
#pragma unroll
        for (int g = 0; g < 4; ++g)
            acc[g] = __builtin_amdgcn_wmma_f32_16x16x32_bf16(
                false, a1.v, false, b1[g].v, (short)0, acc[g], false, false);
    }

    // Lane owns rows b = brow..brow+7 at fixed column n.
    const int brow = btile + (hi16 << 3);

    // Pregates (bf16, contiguous in b thanks to transposed layout)
    float pg[4][8];
#pragma unroll
    for (int g = 0; g < 4; ++g) {
        Pack8 p;
        p.u = *(const u32x4*)(P + ((size_t)((t * 4 + g) * LSTM_H + n)) * LSTM_B + brow);
#pragma unroll
        for (int r = 0; r < 8; ++r) pg[g][r] = (float)p.h[r];
    }

    // Cell state (f32, [n][b] layout => contiguous in b)
    float4 c_lo = *(const float4*)(c_ws + (size_t)n * LSTM_B + brow);
    float4 c_hi = *(const float4*)(c_ws + (size_t)n * LSTM_B + brow + 4);
    float cold[8] = {c_lo.x, c_lo.y, c_lo.z, c_lo.w, c_hi.x, c_hi.y, c_hi.z, c_hi.w};

    const float bI = bhp[n];
    const float bF = bhp[LSTM_H + n];
    const float bG = bhp[2 * LSTM_H + n];
    const float bO = bhp[3 * LSTM_H + n];

    const size_t OUT_H = (size_t)LSTM_B * LSTM_T * LSTM_H;
    const size_t OUT_C = OUT_H + (size_t)LSTM_B * LSTM_H;

    float cnew[8];
#pragma unroll
    for (int r = 0; r < 8; ++r) {
        const float iv = sigm(acc[0][r] + pg[0][r] + bI);
        const float fv = sigm(acc[1][r] + pg[1][r] + bF);
        const float gv = tanhf(acc[2][r] + pg[2][r] + bG);
        const float ov = sigm(acc[3][r] + pg[3][r] + bO);
        const float cn = fv * cold[r] + iv * gv;
        const float hn = ov * tanhf(cn);
        cnew[r] = cn;
        const int b = brow + r;
        // outs[b,t,n]: write-once stream, keep it out of L2
        __builtin_nontemporal_store(hn, out + ((size_t)b * LSTM_T + t) * LSTM_H + n);
        hnext[(size_t)b * LSTM_H + n] = (bf16_t)hn;                // recurrent h
        if (t == LSTM_T - 1) {
            out[OUT_H + (size_t)b * LSTM_H + n] = hn;              // final h
            out[OUT_C + (size_t)b * LSTM_H + n] = cn;              // final c
        }
    }

    float4 s_lo = {cnew[0], cnew[1], cnew[2], cnew[3]};
    float4 s_hi = {cnew[4], cnew[5], cnew[6], cnew[7]};
    *(float4*)(c_ws + (size_t)n * LSTM_B + brow)     = s_lo;
    *(float4*)(c_ws + (size_t)n * LSTM_B + brow + 4) = s_hi;
}

// ---------------------------------------------------------------------------
// Host-side orchestration
// ---------------------------------------------------------------------------
extern "C" void kernel_launch(void* const* d_in, const int* in_sizes, int n_in,
                              void* d_out, int out_size, void* d_ws, size_t ws_size,
                              hipStream_t stream)
{
    // Inputs (dict order): x, then per gate {i,f,g,o}: Wx, bx, Wh, bh
    const float* x   = (const float*)d_in[0];
    const float* Wx[4] = {(const float*)d_in[1], (const float*)d_in[5],
                          (const float*)d_in[9], (const float*)d_in[13]};
    const float* bx[4] = {(const float*)d_in[2], (const float*)d_in[6],
                          (const float*)d_in[10], (const float*)d_in[14]};
    const float* Wh[4] = {(const float*)d_in[3], (const float*)d_in[7],
                          (const float*)d_in[11], (const float*)d_in[15]};
    const float* bh[4] = {(const float*)d_in[4], (const float*)d_in[8],
                          (const float*)d_in[12], (const float*)d_in[16]};
    float* out = (float*)d_out;

    // Workspace carve-up (all sizes multiples of 256B)
    char* ws = (char*)d_ws;
    size_t off = 0;
    bf16_t* Xb   = (bf16_t*)(ws + off); off += (size_t)MTOT * LSTM_D * sizeof(bf16_t);
    bf16_t* Wxp  = (bf16_t*)(ws + off); off += (size_t)NTOT * LSTM_D * sizeof(bf16_t);
    bf16_t* Whp  = (bf16_t*)(ws + off); off += (size_t)NTOT * LSTM_H * sizeof(bf16_t);
    float*  bxp  = (float*) (ws + off); off += (size_t)NTOT * sizeof(float);
    float*  bhp  = (float*) (ws + off); off += (size_t)NTOT * sizeof(float);
    bf16_t* P    = (bf16_t*)(ws + off); off += (size_t)LSTM_T * 4 * LSTM_H * LSTM_B * sizeof(bf16_t);
    bf16_t* h0   = (bf16_t*)(ws + off); off += (size_t)LSTM_B * LSTM_H * sizeof(bf16_t);
    bf16_t* h1   = (bf16_t*)(ws + off); off += (size_t)LSTM_B * LSTM_H * sizeof(bf16_t);
    float*  c_ws = (float*) (ws + off); off += (size_t)LSTM_B * LSTM_H * sizeof(float);
    (void)ws_size; (void)n_in; (void)in_sizes; (void)out_size;

    // --- Phase 0: packing / init ---
    {
        const size_t nthr = (size_t)MTOT * (LSTM_D / 4);
        pack_x_kernel<<<(unsigned)((nthr + 255) / 256), 256, 0, stream>>>(x, Xb);
    }
    {
        const size_t total = (size_t)4 * LSTM_H * LSTM_D;
        pack_w_kernel<<<(unsigned)((total + 255) / 256), 256, 0, stream>>>(
            Wx[0], Wx[1], Wx[2], Wx[3], Wxp, LSTM_D);
        pack_w_kernel<<<(unsigned)((total + 255) / 256), 256, 0, stream>>>(
            Wh[0], Wh[1], Wh[2], Wh[3], Whp, LSTM_H);
    }
    pack_b_kernel<<<(NTOT + 255) / 256, 256, 0, stream>>>(bx[0], bx[1], bx[2], bx[3], bxp);
    pack_b_kernel<<<(NTOT + 255) / 256, 256, 0, stream>>>(bh[0], bh[1], bh[2], bh[3], bhp);
    init_state_kernel<<<(LSTM_B * LSTM_H + 255) / 256, 256, 0, stream>>>(h0, c_ws);

    // --- Phase 1: input projections (big WMMA GEMM) ---
    {
        const int waves  = (MTOT / 16) * (NTOT / 64);   // 196608
        const int blocks = waves / 8;                   // 8 waves (256 thr) per block
        xproj_kernel<<<blocks, 256, 0, stream>>>(Xb, Wxp, bxp, P);
    }

    // --- Phase 2: recurrence, 512 dependent steps (graph-captured) ---
    bf16_t* hp = h0;
    bf16_t* hn = h1;
    for (int t = 0; t < LSTM_T; ++t) {
        lstm_step_kernel<<<LSTM_H / 16, 256, 0, stream>>>(
            hp, Whp, bhp, P, c_ws, hn, out, t);
        bf16_t* tmp = hp; hp = hn; hn = tmp;
    }
}
